// GGMLLayer_86148454023437
// MI455X (gfx1250) — compile-verified, matrix operand order checked by reading, not run on hardware
//
#include <hip/hip_runtime.h>

typedef __attribute__((ext_vector_type(16))) _Float16 v16h;
typedef __attribute__((ext_vector_type(8)))  _Float16 v8h;
typedef __attribute__((ext_vector_type(8)))  float    v8f;
typedef __attribute__((ext_vector_type(4)))  float    v4f;
typedef __attribute__((ext_vector_type(4)))  int      v4i;

#define TILE_M 128
#define TILE_N 128
#define TILE_K 32          // == Q8_0 BLOCK: one scale per (row, k-step)
#define LDSW   40          // padded LDS row stride in halves (80B, 16B-aligned)

// Y[M,N] = X[M,K] * dequant(Q,S)[N,K]^T + bias[N]
__global__ __launch_bounds__(256)
void q8_gemm_wmma(const float* __restrict__ X,
                  const int*   __restrict__ Q,
                  const float* __restrict__ S,
                  const float* __restrict__ Bias,
                  float* __restrict__ Y,
                  int M, int N, int K, int NB) {
    __shared__ __align__(16) _Float16 lA[2][TILE_M * LDSW];
    __shared__ __align__(16) _Float16 lB[2][TILE_N * LDSW];

    const int tid  = threadIdx.x;
    const int lane = tid & 31;
    const int wave = tid >> 5;
    const int wm   = wave & 3;    // wave row  (4 waves x 32 rows)
    const int wn   = wave >> 2;   // wave col  (2 waves x 64 cols)

    const int blockM = blockIdx.y * TILE_M;
    const int blockN = blockIdx.x * TILE_N;

    // loader mapping: each thread fills one 16-element half-row of the 128x32 tile
    const int lrow = tid >> 1;           // 0..127
    const int lcol = (tid & 1) * 16;     // 0 or 16

    // WMMA fragment lane mapping (ISA 7.12.2, wave32, 16-bit A/B)
    const int half_sel = lane >> 4;      // 0: K {0..7,16..23}, 1: K {8..15,24..31}
    const int fm       = lane & 15;      // row (A) / col (B) within 16

    v8f acc[2][4];
    #pragma unroll
    for (int i = 0; i < 2; ++i)
        #pragma unroll
        for (int j = 0; j < 4; ++j)
            acc[i][j] = v8f{};

    // ---------------- software pipeline state ----------------
    v4f   ax[4];     // 16 fp32 of X staged in regs
    v4i   bq[4];     // 16 int8-in-int32 of Q staged in regs
    float bscale;

    const float* aBase = X + (size_t)(blockM + lrow) * K + lcol;
    const int*   bBase = Q + (size_t)(blockN + lrow) * K + lcol;
    const float* sBase = S + (size_t)(blockN + lrow) * NB;

    auto load_global = [&](int k0) {
        #pragma unroll
        for (int v = 0; v < 4; ++v) ax[v] = *(const v4f*)(aBase + k0 + v * 4);
        bscale = sBase[k0 >> 5];
        #pragma unroll
        for (int v = 0; v < 4; ++v) bq[v] = *(const v4i*)(bBase + k0 + v * 4);
    };

    auto store_lds = [&](int buf) {
        _Float16* dA = &lA[buf][lrow * LDSW + lcol];
        v8h a0, a1;
        #pragma unroll
        for (int v = 0; v < 2; ++v)
            #pragma unroll
            for (int e = 0; e < 4; ++e) {
                a0[v * 4 + e] = (_Float16)ax[v][e];
                a1[v * 4 + e] = (_Float16)ax[2 + v][e];
            }
        *(v8h*)dA       = a0;
        *(v8h*)(dA + 8) = a1;

        _Float16* dB = &lB[buf][lrow * LDSW + lcol];
        v8h b0, b1;
        #pragma unroll
        for (int v = 0; v < 2; ++v)
            #pragma unroll
            for (int e = 0; e < 4; ++e) {
                b0[v * 4 + e] = (_Float16)((float)bq[v][e]     * bscale);
                b1[v * 4 + e] = (_Float16)((float)bq[2 + v][e] * bscale);
            }
        *(v8h*)dB       = b0;
        *(v8h*)(dB + 8) = b1;
    };

    auto compute = [&](int buf) {
        v16h a[2], b[4];
        #pragma unroll
        for (int i = 0; i < 2; ++i) {
            const _Float16* pa = &lA[buf][(wm * 32 + i * 16 + fm) * LDSW + half_sel * 8];
            v8h lo = *(const v8h*)pa;
            v8h hi = *(const v8h*)(pa + 16);
            #pragma unroll
            for (int e = 0; e < 8; ++e) { a[i][e] = lo[e]; a[i][e + 8] = hi[e]; }
        }
        #pragma unroll
        for (int j = 0; j < 4; ++j) {
            const _Float16* pb = &lB[buf][(wn * 64 + j * 16 + fm) * LDSW + half_sel * 8];
            v8h lo = *(const v8h*)pb;
            v8h hi = *(const v8h*)(pb + 16);
            #pragma unroll
            for (int e = 0; e < 8; ++e) { b[j][e] = lo[e]; b[j][e + 8] = hi[e]; }
        }
        #pragma unroll
        for (int i = 0; i < 2; ++i)
            #pragma unroll
            for (int j = 0; j < 4; ++j)
                acc[i][j] = __builtin_amdgcn_wmma_f32_16x16x32_f16(
                    /*neg_a=*/false, a[i], /*neg_b=*/false, b[j],
                    /*c_mod=*/(short)0, acc[i][j],
                    /*reuse_a=*/false, /*reuse_b=*/false);
    };

    // ---------------- main loop: double-buffered, 1 barrier/step ----------------
    const int steps = K / TILE_K;

    load_global(0);
    store_lds(0);
    __syncthreads();

    for (int k = 0; k < steps - 1; ++k) {
        load_global((k + 1) * TILE_K);   // overlap global fetch with WMMAs
        compute(k & 1);
        store_lds((k + 1) & 1);          // other buffer: no pre-barrier needed
        __syncthreads();
    }
    compute((steps - 1) & 1);

    // ---- epilogue: C/D layout (lane<16: M=r, lane>=16: M=8+r; N=lane&15) ----
    // 32-bit linear indexing: row*N+col < 2^24, avoids v_mul_u64 chains.
    #pragma unroll
    for (int j = 0; j < 4; ++j) {
        const uint32_t col = (uint32_t)(blockN + wn * 64 + j * 16 + fm);
        const float bv = Bias[col];
        #pragma unroll
        for (int i = 0; i < 2; ++i) {
            const uint32_t row0 = (uint32_t)(blockM + wm * 32 + i * 16 + 8 * half_sel);
            #pragma unroll
            for (int r = 0; r < 8; ++r) {
                const uint32_t idx = (row0 + (uint32_t)r) * (uint32_t)N + col;
                Y[idx] = acc[i][j][r] + bv;
            }
        }
    }
}

extern "C" void kernel_launch(void* const* d_in, const int* in_sizes, int n_in,
                              void* d_out, int out_size, void* d_ws, size_t ws_size,
                              hipStream_t stream) {
    const float* X    = (const float*)d_in[0];   // [B,S,I] fp32
    const int*   Q    = (const int*)  d_in[1];   // [O,I] int8 values in int32
    const float* S    = (const float*)d_in[2];   // [O,NB] fp32
    const float* Bias = (const float*)d_in[3];   // [O] fp32
    float*       Y    = (float*)d_out;           // [B,S,O] fp32

    const int I  = 4096;
    const int O  = 4096;
    const int NB = I / 32;
    const int M  = in_sizes[0] / I;              // B*S = 4096

    dim3 grid(O / TILE_N, M / TILE_M);
    q8_gemm_wmma<<<grid, 256, 0, stream>>>(X, Q, S, Bias, Y, M, O, I, NB);
}
